// SimpleFeatureMLP_77532749627706
// MI455X (gfx1250) — matrix operand order repaired and checked by reading, not compiled, not run on hardware
//
#include <hip/hip_runtime.h>
#include <hip/hip_bf16.h>

typedef unsigned int u32;
typedef unsigned short u16;
typedef __attribute__((ext_vector_type(16))) __bf16 v16bf;
typedef __attribute__((ext_vector_type(8)))  float  v8f;
typedef __attribute__((ext_vector_type(4)))  u32    v4u;

#define B_ROWS   131072
#define IN_PAD   128      // 119 padded to 128
#define HIDDEN   1024
#define OUT_N    256
#define HSTRIDE  1032     // 1024 + 8 halves pad: 16B-aligned rows, conflict-free b128 reads

union BF16x16 { v16bf v; v4u u[2]; };

__device__ inline u16 f2bf(float f) {
    u32 u = __float_as_uint(f);
    u32 r = (u + 0x7FFFu + ((u >> 16) & 1u)) >> 16;
    return (u16)r;
}
__device__ inline u32 pack2(float lo, float hi) {
    return (u32)f2bf(lo) | ((u32)f2bf(hi) << 16);
}
__device__ inline v8f vzero8() {
    v8f z = {0.f,0.f,0.f,0.f,0.f,0.f,0.f,0.f};
    return z;
}
__device__ inline v8f wmma_bf16(const v16bf& a, const v16bf& b, const v8f& c) {
    return __builtin_amdgcn_wmma_f32_16x16x32_bf16(false, a, false, b, (short)0, c,
                                                   false, false);
}

// ---------------------------------------------------------------------------
// Kernel 1: per-row poker features + bf16 row packing into X[B][128]
// ---------------------------------------------------------------------------
__global__ __launch_bounds__(256) void feat_pack_kernel(
    const float* __restrict__ hole, const float* __restrict__ board,
    u16* __restrict__ X)
{
    int i = blockIdx.x * blockDim.x + threadIdx.x;
    const float4* h4 = (const float4*)(hole  + (size_t)i * 52);
    const float4* b4 = (const float4*)(board + (size_t)i * 52);

    float4 hr[13], br[13];
    float brc[13], arc[13];
    float bsc[4] = {0,0,0,0}, asc[4] = {0,0,0,0};
    float bcount = 0.f, hcount = 0.f;
    #pragma unroll
    for (int r = 0; r < 13; ++r) {
        hr[r] = h4[r]; br[r] = b4[r];
        float rb = br[r].x + br[r].y + br[r].z + br[r].w;
        float rh = hr[r].x + hr[r].y + hr[r].z + hr[r].w;
        brc[r] = rb; arc[r] = rb + rh;
        bcount += rb; hcount += rh;
        bsc[0] += br[r].x; bsc[1] += br[r].y; bsc[2] += br[r].z; bsc[3] += br[r].w;
        asc[0] += br[r].x + hr[r].x; asc[1] += br[r].y + hr[r].y;
        asc[2] += br[r].z + hr[r].z; asc[3] += br[r].w + hr[r].w;
    }

    int pairs_b = 0; bool trips_b = false;
    #pragma unroll
    for (int r = 0; r < 13; ++r) { pairs_b += (brc[r] >= 2.f); trips_b |= (brc[r] >= 3.f); }
    float maxbsc = fmaxf(fmaxf(bsc[0], bsc[1]), fmaxf(bsc[2], bsc[3]));

    float pp_b[17], pp_a[17];
    #pragma unroll
    for (int r = 0; r < 13; ++r) { pp_b[r] = (brc[r] > 0.f) ? 1.f : 0.f; pp_a[r] = (arc[r] > 0.f) ? 1.f : 0.f; }
    #pragma unroll
    for (int r = 13; r < 17; ++r) { pp_b[r] = 0.f; pp_a[r] = 0.f; }

    bool straight_b_any = false;
    #pragma unroll
    for (int r = 0; r < 13; ++r) {
        float w = pp_b[r] + pp_b[r+1] + pp_b[r+2] + pp_b[r+3] + pp_b[r+4];
        straight_b_any |= (w >= 3.f);
    }

    float strength = trips_b ? 0.8f : (pairs_b >= 2 ? 0.6f : (pairs_b >= 1 ? 0.4f : 0.2f));
    if (!(bcount > 0.f)) strength = 0.f;
    float flush_bf    = (maxbsc >= 3.f && bcount > 0.f) ? 1.f : 0.f;
    float straight_bf = (straight_b_any && bcount > 0.f) ? 1.f : 0.f;
    float gr0 = (bcount == 0.f) ? 1.f : 0.f;
    float gr1 = (bcount == 3.f) ? 1.f : 0.f;
    float gr2 = (bcount == 4.f) ? 1.f : 0.f;
    float gr3 = (bcount == 5.f) ? 1.f : 0.f;

    float validf = (hcount >= 2.f && bcount >= 1.f) ? 1.f : 0.f;
    float msc = fmaxf(fmaxf(asc[0], asc[1]), fmaxf(asc[2], asc[3]));
    float flush_draw = (msc == 4.f) ? 1.f : 0.f;
    float flush_outs = fmaxf(0.f, 13.f - msc) * (1.f / 13.f);

    bool found = false; float c4_first = 0.f;
    #pragma unroll
    for (int r = 0; r < 13; ++r) {
        float c5 = pp_a[r] + pp_a[r+1] + pp_a[r+2] + pp_a[r+3] + pp_a[r+4];
        float c4 = pp_a[r] + pp_a[r+1] + pp_a[r+2] + pp_a[r+3];
        bool q = (pp_a[r] > 0.f) && (c5 >= 4.f);
        if (q && !found) { found = true; c4_first = c4; }
        if (r == 0 && !found) c4_first = c4;   // argmax(all false)==0 -> c4[0]
    }
    float straight_drawf = found ? 1.f : 0.f;
    float straight_outs = straight_drawf * ((c4_first >= 4.f) ? 0.4f : 0.2f);
    float total_outs = flush_draw * flush_outs * 9.f + straight_drawf * straight_outs * 8.f;
    float remaining = 52.f - (hcount + bcount);
    float equity = (remaining > 0.f) ? fminf(1.f, total_outs / fmaxf(remaining, 1.f)) : 0.f;

    int pairs_a = 0; bool trips_a = false;
    #pragma unroll
    for (int r = 0; r < 13; ++r) { pairs_a += (arc[r] >= 2.f); trips_a |= (arc[r] >= 3.f); }
    float hit_pair  = (pairs_a >= 1) ? 1.f : 0.f;
    float hit_trips = trips_a ? 1.f : 0.f;
    float hit_two   = (pairs_a >= 2) ? 1.f : 0.f;

    float f[16] = { strength, flush_bf, straight_bf, gr0, gr1, gr2, gr3,
                    validf * flush_draw, validf * flush_outs,
                    validf * straight_drawf, validf * straight_outs, validf * equity,
                    validf * hit_pair, validf * hit_trips, validf * hit_two, 0.f };

    u32* Xo = (u32*)X + (size_t)i * (IN_PAD / 2);
    #pragma unroll
    for (int r = 0; r < 13; ++r) {
        Xo[2*r]      = pack2(hr[r].x, hr[r].y);
        Xo[2*r + 1]  = pack2(hr[r].z, hr[r].w);
        Xo[26 + 2*r] = pack2(br[r].x, br[r].y);
        Xo[27 + 2*r] = pack2(br[r].z, br[r].w);
    }
    #pragma unroll
    for (int j = 0; j < 8; ++j) Xo[52 + j] = pack2(f[2*j], f[2*j + 1]);
    #pragma unroll
    for (int j = 60; j < 64; ++j) Xo[j] = 0u;
}

// ---------------------------------------------------------------------------
// Weight pre-swizzle into the WMMA bf16 B-operand layout:
//   per (kBlock32, nTile16): 256 dwords, dword = lane*8 + j,
//   lane half h: dword j holds K=8h+2j,+1 (j<4), K=16+8h+2(j-4),+1 (j>=4)
// ---------------------------------------------------------------------------
__global__ __launch_bounds__(256) void pack_w1_kernel(
    const float* __restrict__ W1, u32* __restrict__ W1p)
{
    int tid = blockIdx.x * blockDim.x + threadIdx.x;     // 4*64*256 = 65536
    int kb  = tid >> 14;
    int rem = tid & 16383;
    int nb  = rem >> 8;
    int t   = rem & 255;
    int lane = t >> 3, j = t & 7;
    int k = kb * 32 + ((lane >> 4) << 3) + ((j >> 2) << 4) + 2 * (j & 3);
    int n = nb * 16 + (lane & 15);
    float lo = (k     < 119) ? W1[k * HIDDEN + n]       : 0.f;
    float hi = (k + 1 < 119) ? W1[(k + 1) * HIDDEN + n] : 0.f;
    W1p[tid] = pack2(lo, hi);
}

__global__ __launch_bounds__(256) void pack_w2_kernel(
    const float* __restrict__ W2, u32* __restrict__ W2p)
{
    int tid = blockIdx.x * blockDim.x + threadIdx.x;     // 32*16*256 = 131072
    int kb  = tid >> 12;
    int rem = tid & 4095;
    int nb  = rem >> 8;
    int t   = rem & 255;
    int lane = t >> 3, j = t & 7;
    int k = kb * 32 + ((lane >> 4) << 3) + ((j >> 2) << 4) + 2 * (j & 3);
    int n = nb * 16 + (lane & 15);
    W2p[tid] = pack2(W2[k * OUT_N + n], W2[(k + 1) * OUT_N + n]);
}

// ---------------------------------------------------------------------------
// Fused MLP: 64 rows per 256-thread block (8 waves), H staged in LDS (bf16)
// ---------------------------------------------------------------------------
__global__ __launch_bounds__(256) void mlp_kernel(
    const u16* __restrict__ X,
    const u32* __restrict__ W1p, const u32* __restrict__ W2p,
    const float* __restrict__ b1, const float* __restrict__ b2,
    float* __restrict__ out)
{
    __shared__ __bf16 Hs[64 * HSTRIDE];   // 132,096 bytes (CDNA5: 320KB LDS/WGP)

    const int lane    = threadIdx.x & 31;
    const int wv      = threadIdx.x >> 5;   // 0..7
    const int ln15    = lane & 15;
    const int lhalf   = lane >> 4;          // 0/1 (K-half)
    const int rowTile = wv & 3;             // 0..3 : 16-row tile
    const int grp     = wv >> 2;            // 0/1
    const int blockRow = blockIdx.x * 64;

    // ---- GEMM1: X[64x128] @ W1[128x1024] -> relu -> Hs (bf16) ----
    BF16x16 a[4];
    {
        const u16* xr = X + (size_t)(blockRow + rowTile * 16 + ln15) * IN_PAD;
        #pragma unroll
        for (int kb = 0; kb < 4; ++kb) {
            int off = kb * 32 + lhalf * 8;
            a[kb].u[0] = *(const v4u*)(xr + off);
            a[kb].u[1] = *(const v4u*)(xr + off + 16);
        }
    }

    const int row0h = rowTile * 16 + 8 * lhalf;
    for (int j = 0; j < 16; ++j) {
        // two independent accumulator chains per iteration
        const int nT0 = 4 * j + 2 * grp;     // grp0: 0,1,4,5,...  grp1: 2,3,6,7,...
        const int nT1 = nT0 + 1;
        v8f c0 = vzero8(), c1 = vzero8();
        #pragma unroll
        for (int kb = 0; kb < 4; ++kb) {
            const u32* bp0 = W1p + ((kb * 64 + nT0) * 256) + lane * 8;
            BF16x16 bm0, bm1;
            bm0.u[0] = *(const v4u*)(bp0);
            bm0.u[1] = *(const v4u*)(bp0 + 4);
            bm1.u[0] = *(const v4u*)(bp0 + 256);
            bm1.u[1] = *(const v4u*)(bp0 + 260);
            c0 = wmma_bf16(a[kb].v, bm0.v, c0);
            c1 = wmma_bf16(a[kb].v, bm1.v, c1);
        }
        {
            const int col = nT0 * 16 + ln15;
            const float bias = b1[col];
            #pragma unroll
            for (int r = 0; r < 8; ++r)
                Hs[(row0h + r) * HSTRIDE + col] = (__bf16)fmaxf(c0[r] + bias, 0.f);
        }
        {
            const int col = nT1 * 16 + ln15;
            const float bias = b1[col];
            #pragma unroll
            for (int r = 0; r < 8; ++r)
                Hs[(row0h + r) * HSTRIDE + col] = (__bf16)fmaxf(c1[r] + bias, 0.f);
        }
    }
    __syncthreads();

    // ---- GEMM2: Hs[64x1024] @ W2[1024x256] + b2 -> out ----
    v8f acc[8];
    #pragma unroll
    for (int t = 0; t < 8; ++t) acc[t] = vzero8();

    const int hbase = (rowTile * 16 + ln15) * HSTRIDE;
    for (int kb = 0; kb < 32; ++kb) {
        BF16x16 a2;
        const int off = hbase + kb * 32 + lhalf * 8;
        a2.u[0] = *(const v4u*)(&Hs[off]);
        a2.u[1] = *(const v4u*)(&Hs[off + 16]);
        #pragma unroll
        for (int t = 0; t < 8; ++t) {
            const int nT = grp * 8 + t;   // 0..15
            const u32* bp = W2p + ((kb * 16 + nT) * 256) + lane * 8;
            BF16x16 bm;
            bm.u[0] = *(const v4u*)(bp);
            bm.u[1] = *(const v4u*)(bp + 4);
            acc[t] = wmma_bf16(a2.v, bm.v, acc[t]);
        }
    }
    #pragma unroll
    for (int t = 0; t < 8; ++t) {
        const int   nT   = grp * 8 + t;
        const int   col  = nT * 16 + ln15;
        const float bias = b2[col];
        const int   row0 = blockRow + rowTile * 16 + 8 * lhalf;
        #pragma unroll
        for (int r = 0; r < 8; ++r)
            out[(size_t)(row0 + r) * OUT_N + col] = acc[t][r] + bias;
    }
}

// ---------------------------------------------------------------------------
extern "C" void kernel_launch(void* const* d_in, const int* in_sizes, int n_in,
                              void* d_out, int out_size, void* d_ws, size_t ws_size,
                              hipStream_t stream) {
    const float* hole  = (const float*)d_in[0];
    const float* board = (const float*)d_in[1];
    const float* W1    = (const float*)d_in[2];
    const float* b1    = (const float*)d_in[3];
    const float* W2    = (const float*)d_in[4];
    const float* b2    = (const float*)d_in[5];
    float* out = (float*)d_out;

    // Workspace layout: X[B][128] bf16 | W1p (64K dwords) | W2p (128K dwords)
    u16* X   = (u16*)d_ws;
    u32* W1p = (u32*)((char*)d_ws + (size_t)B_ROWS * IN_PAD * sizeof(u16));
    u32* W2p = W1p + 65536;

    feat_pack_kernel<<<B_ROWS / 256, 256, 0, stream>>>(hole, board, X);
    pack_w1_kernel<<<65536 / 256, 256, 0, stream>>>(W1, W1p);
    pack_w2_kernel<<<131072 / 256, 256, 0, stream>>>(W2, W2p);
    mlp_kernel<<<B_ROWS / 64, 256, 0, stream>>>(X, W1p, W2p, b1, b2, out);
}
